// LongTermAccompanimentBeatwiseUpcomingBars_3487513444820
// MI455X (gfx1250) — matrix operand order repaired and checked by reading, compiled, not run on hardware
//
#include <hip/hip_runtime.h>
#include <hip/hip_bf16.h>
#include <math.h>

#define B_ 16
#define T_ 512
#define S_ 256
#define D_ 512
#define NH_ 8
#define HD_ 64
#define FF_ 2048
#define L_ 6
#define V_ 9

typedef __attribute__((ext_vector_type(16))) __bf16 v16bf;
typedef __attribute__((ext_vector_type(8)))  float  v8f;

union FragAB {
  v16bf v;
  unsigned int u[8];
  unsigned short s[16];
};

__device__ __forceinline__ unsigned short f2bf(float f) {
  union { float f; unsigned u; } a; a.f = f;
  unsigned u = a.u;
  unsigned r = u + 0x7FFFu + ((u >> 16) & 1u);   // round-to-nearest-even
  return (unsigned short)(r >> 16);
}

// CDNA5 async global->LDS copy (ASYNCcnt-tracked), 16 bytes per lane.
__device__ __forceinline__ void async_ld128(unsigned lds_off, unsigned long long gaddr) {
  asm volatile("global_load_async_to_lds_b128 %0, %1, off"
               :: "v"(lds_off), "v"(gaddr) : "memory");
}
__device__ __forceinline__ void wait_async0() {
  asm volatile("s_wait_asynccnt 0x0" ::: "memory");
}
__device__ __forceinline__ unsigned lds_addr32(const void* p) {
  // flat->LDS uses addr[31:0] (ISA 10.2), i.e. the workgroup LDS byte offset
  return (unsigned)(unsigned long long)p;
}

// ---------------------------------------------------------------------------
// f32 -> bf16 bulk conversion (weights / memory, once per launch)
// ---------------------------------------------------------------------------
__global__ __launch_bounds__(256) void k_cvt(
    const float* __restrict__ in, unsigned short* __restrict__ out, int n)
{
  int i = blockIdx.x * 256 + threadIdx.x;
  if (i < n) out[i] = f2bf(in[i]);
}

// ---------------------------------------------------------------------------
// Embedding: x = tgt @ emb_w^T + emb_b + PE(t, d); emits f32 + bf16 copies
// ---------------------------------------------------------------------------
__global__ __launch_bounds__(256) void k_embed(
    const float* __restrict__ tgt, const float* __restrict__ ew,
    const float* __restrict__ eb, float* __restrict__ X,
    unsigned short* __restrict__ Xb)
{
  int i = blockIdx.x * 256 + threadIdx.x;
  if (i >= B_ * T_ * D_) return;
  int d  = i & (D_ - 1);
  int bt = i >> 9;
  int t  = bt & (T_ - 1);
  const float* trow = tgt + (size_t)bt * (3 * V_);
  float acc = eb[d];
#pragma unroll
  for (int j = 0; j < 3 * V_; j++) acc += trow[j] * ew[d * (3 * V_) + j];
  int twoI = d & ~1;
  float freq = __expf(-0.0179889463f * (float)twoI);   // ln(10000)/512
  float ang  = (float)t * freq;
  float pe   = (d & 1) ? __cosf(ang) : __sinf(ang);
  float r = acc + pe;
  X[i]  = r;
  Xb[i] = f2bf(r);
}

// ---------------------------------------------------------------------------
// WMMA bf16 GEMM, async double-buffered LDS pipeline.
// C[M,N] = act(A[M,K] @ W[N,K]^T + bias[N]);  A,W bf16; out f32 or bf16.
// Block tile 128(M) x 64(N), BK=32, 8 waves, each wave 32x32 (2x2 WMMA accs).
// ---------------------------------------------------------------------------
__global__ __launch_bounds__(256) void k_gemm_bias(
    const unsigned short* __restrict__ A, const unsigned short* __restrict__ W,
    const float* __restrict__ bias, float* __restrict__ Cf,
    unsigned short* __restrict__ Cb,
    int M, int N, int K, int relu, int outBf)
{
  // 80B pitches: 16B-aligned for B128 async writes, conflict-free frag reads
  __shared__ __attribute__((aligned(16))) unsigned short sA[2][128][40];
  __shared__ __attribute__((aligned(16))) unsigned short sB[2][64][40];
  const int tid   = threadIdx.x;
  const int lane  = tid & 31;
  const int wave  = tid >> 5;
  const int half  = lane >> 4;
  const int l16   = lane & 15;
  const int waveM = wave & 3;
  const int waveN = wave >> 2;
  const int blockM = blockIdx.y * 128;
  const int blockN = blockIdx.x * 64;

  v8f acc[2][2];
#pragma unroll
  for (int i = 0; i < 2; i++)
#pragma unroll
    for (int j = 0; j < 2; j++)
#pragma unroll
      for (int e = 0; e < 8; e++) acc[i][j][e] = 0.f;

  const int nT = K >> 5;

  // issue async copies for K-tile (k0) into buffer `buf`
  auto issue_tile = [&](int buf, int k0) {
    unsigned sA0 = lds_addr32(&sA[buf][0][0]);
    unsigned sB0 = lds_addr32(&sB[buf][0][0]);
#pragma unroll
    for (int t2 = 0; t2 < 2; t2++) {            // A: 512 x 16B chunks
      int idx = tid + t2 * 256;
      int r = idx >> 2, cc = idx & 3;
      async_ld128(sA0 + r * 80 + cc * 16,
                  (unsigned long long)(A + (size_t)(blockM + r) * K + k0 + cc * 8));
    }
    {                                            // W: 256 x 16B chunks
      int r = tid >> 2, cc = tid & 3;
      async_ld128(sB0 + r * 80 + cc * 16,
                  (unsigned long long)(W + (size_t)(blockN + r) * K + k0 + cc * 8));
    }
  };

  issue_tile(0, 0);
  for (int i = 0; i < nT; i++) {
    wait_async0();
    __syncthreads();                      // tile i resident; buf i^1 free everywhere
    if (i + 1 < nT) issue_tile((i + 1) & 1, (i + 1) << 5);
    const int bs = i & 1;

    FragAB af[2], bf[2];
#pragma unroll
    for (int mt = 0; mt < 2; mt++) {
      int mr = waveM * 32 + mt * 16 + l16;
#pragma unroll
      for (int j = 0; j < 8; j++) {
        int kb = (j < 4) ? (half * 8 + 2 * j) : (16 + half * 8 + 2 * (j - 4));
        af[mt].u[j] = *(const unsigned int*)&sA[bs][mr][kb];
      }
    }
#pragma unroll
    for (int nt = 0; nt < 2; nt++) {
      int nr = waveN * 32 + nt * 16 + l16;
#pragma unroll
      for (int j = 0; j < 8; j++)
        bf[nt].u[j] = *(const unsigned int*)&sB[bs][nr][(j < 4) ? (half * 8 + 2 * j) : (16 + half * 8 + 2 * (j - 4))];
    }
#pragma unroll
    for (int mt = 0; mt < 2; mt++)
#pragma unroll
      for (int nt = 0; nt < 2; nt++)
        acc[mt][nt] = __builtin_amdgcn_wmma_f32_16x16x32_bf16(
            false, af[mt].v, false, bf[nt].v, (short)0, acc[mt][nt], false, false);
  }

#pragma unroll
  for (int nt = 0; nt < 2; nt++) {
    int ng = blockN + waveN * 32 + nt * 16 + l16;
    float bn = bias[ng];
#pragma unroll
    for (int mt = 0; mt < 2; mt++) {
#pragma unroll
      for (int j = 0; j < 8; j++) {
        int mg = blockM + waveM * 32 + mt * 16 + j + 8 * half;
        float v = acc[mt][nt][j] + bn;
        if (relu) v = fmaxf(v, 0.f);
        if (outBf) Cb[(size_t)mg * N + ng] = f2bf(v);
        else       Cf[(size_t)mg * N + ng] = v;
      }
    }
  }
}

// ---------------------------------------------------------------------------
// Fused flash attention, bf16 in / bf16 out. One wave per (b, h, 32 q-rows).
// mode 0: causal.  mode 1: banded memory mask (col >= max(1, t>>4)).
// ---------------------------------------------------------------------------
__global__ __launch_bounds__(32) void k_flash_attn(
    const unsigned short* __restrict__ Q, const unsigned short* __restrict__ Kb,
    const unsigned short* __restrict__ Vb, unsigned short* __restrict__ O,
    int lenq, int lenkv, int mode)
{
  __shared__ __attribute__((aligned(16))) unsigned short sQ[32][72];
  __shared__ __attribute__((aligned(16))) unsigned short sK[32][72];
  __shared__ __attribute__((aligned(16))) unsigned short sV[32][72];
  __shared__ float          sS[32][33];
  __shared__ unsigned short sP[32][34];
  __shared__ float          sCorr[32];
  __shared__ float          sL[32];

  const int lane = threadIdx.x;
  const int half = lane >> 4;
  const int l16  = lane & 15;
  const int b    = blockIdx.x / NH_;
  const int h    = blockIdx.x % NH_;
  const int t0   = blockIdx.y * 32;

  // Stage Q tile (32 x 64 bf16 = 256 x 16B) via async copy
  {
    unsigned sQ0 = lds_addr32(&sQ[0][0]);
#pragma unroll
    for (int i = 0; i < 8; i++) {
      int idx = lane + i * 32;
      int r = idx >> 3, cc = idx & 7;
      async_ld128(sQ0 + r * 144 + cc * 16,
                  (unsigned long long)(Q + (size_t)(b * lenq + t0 + r) * D_ + h * HD_ + cc * 8));
    }
    wait_async0();
  }
  __syncthreads();

  FragAB qf[2][2];
#pragma unroll
  for (int mt = 0; mt < 2; mt++) {
    int mr = mt * 16 + l16;
#pragma unroll
    for (int kk = 0; kk < 2; kk++)
#pragma unroll
      for (int j = 0; j < 8; j++) {
        int kb = kk * 32 + ((j < 4) ? (half * 8 + 2 * j) : (16 + half * 8 + 2 * (j - 4)));
        qf[mt][kk].u[j] = *(const unsigned int*)&sQ[mr][kb];
      }
  }

  v8f o[2][4];
#pragma unroll
  for (int mt = 0; mt < 2; mt++)
#pragma unroll
    for (int nt = 0; nt < 4; nt++)
#pragma unroll
      for (int e = 0; e < 8; e++) o[mt][nt][e] = 0.f;

  float m_i = -1e30f, l_i = 0.f;

  int kvEnd;
  if (mode == 0) {
    kvEnd = t0 + 32; if (kvEnd > lenkv) kvEnd = lenkv;      // causal clamp
  } else {
    int allow = (t0 + 31) >> 4;
    if (allow <= 1) allow = 1;
    kvEnd = (allow < lenkv) ? allow : lenkv;                 // banded clamp
  }

  for (int kc = 0; kc < kvEnd; kc += 32) {
    {
      unsigned sK0 = lds_addr32(&sK[0][0]);
      unsigned sV0 = lds_addr32(&sV[0][0]);
#pragma unroll
      for (int i = 0; i < 8; i++) {
        int idx = lane + i * 32;
        int r = idx >> 3, cc = idx & 7;
        size_t gb = (size_t)(b * lenkv + kc + r) * D_ + h * HD_ + cc * 8;
        async_ld128(sK0 + r * 144 + cc * 16, (unsigned long long)(Kb + gb));
        async_ld128(sV0 + r * 144 + cc * 16, (unsigned long long)(Vb + gb));
      }
      wait_async0();
    }
    __syncthreads();

    // S = (Q . K^T) * scale, masked, into sS
#pragma unroll
    for (int mt = 0; mt < 2; mt++)
#pragma unroll
      for (int nt = 0; nt < 2; nt++) {
        v8f s;
#pragma unroll
        for (int e = 0; e < 8; e++) s[e] = 0.f;
#pragma unroll
        for (int kk = 0; kk < 2; kk++) {
          FragAB kf;
          int nr = nt * 16 + l16;
#pragma unroll
          for (int j = 0; j < 8; j++)
            kf.u[j] = *(const unsigned int*)&sK[nr][kk * 32 + ((j < 4) ? (half * 8 + 2 * j) : (16 + half * 8 + 2 * (j - 4)))];
          s = __builtin_amdgcn_wmma_f32_16x16x32_bf16(
              false, qf[mt][kk].v, false, kf.v, (short)0, s, false, false);
        }
#pragma unroll
        for (int j = 0; j < 8; j++) {
          int m = mt * 16 + j + 8 * half;
          int n = nt * 16 + l16;
          int gq = t0 + m, gc = kc + n;
          bool masked;
          if (mode == 0) {
            masked = gc > gq;
          } else {
            int allow = gq >> 4;
            if (allow <= 1) allow = 1;
            masked = gc >= allow;
          }
          sS[m][n] = masked ? -1e30f : s[j] * 0.125f;   // 1/sqrt(64)
        }
      }
    __syncthreads();

    // Online softmax: lane r owns row r
    {
      float tm = -1e30f;
#pragma unroll
      for (int c = 0; c < 32; c++) tm = fmaxf(tm, sS[lane][c]);
      float mnew = fmaxf(m_i, tm);
      float corr = __expf(m_i - mnew);
      l_i *= corr;
#pragma unroll
      for (int c = 0; c < 32; c++) {
        float p = __expf(sS[lane][c] - mnew);
        l_i += p;
        sP[lane][c] = f2bf(p);
      }
      m_i = mnew;
      sCorr[lane] = corr;
    }
    __syncthreads();

#pragma unroll
    for (int mt = 0; mt < 2; mt++)
#pragma unroll
      for (int j = 0; j < 8; j++) {
        float cf = sCorr[mt * 16 + j + 8 * half];
#pragma unroll
        for (int nt = 0; nt < 4; nt++) o[mt][nt][j] *= cf;
      }

    // O += P . V
    FragAB pf[2];
#pragma unroll
    for (int mt = 0; mt < 2; mt++) {
      int mr = mt * 16 + l16;
#pragma unroll
      for (int j = 0; j < 8; j++) {
        int kb = (j < 4) ? (half * 8 + 2 * j) : (16 + half * 8 + 2 * (j - 4));
        pf[mt].u[j] = *(const unsigned int*)&sP[mr][kb];
      }
    }
#pragma unroll
    for (int nt = 0; nt < 4; nt++) {
      FragAB vf;
      int nc = nt * 16 + l16;
#pragma unroll
      for (int e = 0; e < 16; e++) vf.s[e] = sV[(e < 8) ? (half * 8 + e) : (16 + half * 8 + (e - 8))][nc];
#pragma unroll
      for (int mt = 0; mt < 2; mt++)
        o[mt][nt] = __builtin_amdgcn_wmma_f32_16x16x32_bf16(
            false, pf[mt].v, false, vf.v, (short)0, o[mt][nt], false, false);
    }
    __syncthreads();
  }

  sL[lane] = l_i;
  __syncthreads();
#pragma unroll
  for (int mt = 0; mt < 2; mt++)
#pragma unroll
    for (int j = 0; j < 8; j++) {
      int m = mt * 16 + j + 8 * half;
      float inv = 1.f / sL[m];
#pragma unroll
      for (int nt = 0; nt < 4; nt++) {
        int n = nt * 16 + l16;
        O[(size_t)(b * lenq + t0 + m) * D_ + h * HD_ + n] = f2bf(o[mt][nt][j] * inv);
      }
    }
}

// ---------------------------------------------------------------------------
// Out = LayerNorm(X [+ R]) * gamma + beta; emits f32 + bf16 copies.
// ---------------------------------------------------------------------------
__global__ __launch_bounds__(256) void k_add_ln(
    const float* __restrict__ X, const float* __restrict__ R,
    const float* __restrict__ gamma, const float* __restrict__ beta,
    float* __restrict__ Out, unsigned short* __restrict__ OutB, int hasRes)
{
  __shared__ float red[256];
  int row = blockIdx.x;
  int tid = threadIdx.x;
  size_t base = (size_t)row * D_;
  float v0 = X[base + tid];
  float v1 = X[base + tid + 256];
  if (hasRes) { v0 += R[base + tid]; v1 += R[base + tid + 256]; }
  red[tid] = v0 + v1;
  __syncthreads();
  for (int s2 = 128; s2 > 0; s2 >>= 1) {
    if (tid < s2) red[tid] += red[tid + s2];
    __syncthreads();
  }
  float mean = red[0] * (1.f / (float)D_);
  __syncthreads();
  float d0 = v0 - mean, d1 = v1 - mean;
  red[tid] = d0 * d0 + d1 * d1;
  __syncthreads();
  for (int s2 = 128; s2 > 0; s2 >>= 1) {
    if (tid < s2) red[tid] += red[tid + s2];
    __syncthreads();
  }
  float rs = rsqrtf(red[0] * (1.f / (float)D_) + 1e-5f);
  float y0 = d0 * rs * gamma[tid]       + beta[tid];
  float y1 = d1 * rs * gamma[tid + 256] + beta[tid + 256];
  Out[base + tid]        = y0;
  Out[base + tid + 256]  = y1;
  OutB[base + tid]       = f2bf(y0);
  OutB[base + tid + 256] = f2bf(y1);
}

// ---------------------------------------------------------------------------
// Output heads: three (V=9, D=512) linears, outputs concatenated.
// ---------------------------------------------------------------------------
__global__ __launch_bounds__(128) void k_heads(
    const float* __restrict__ X,
    const float* __restrict__ hw, const float* __restrict__ hb,
    const float* __restrict__ vw, const float* __restrict__ vb,
    const float* __restrict__ ow, const float* __restrict__ ob,
    float* __restrict__ out)
{
  __shared__ float sx[D_];
  int row = blockIdx.x;
  int tid = threadIdx.x;
  for (int c = tid; c < D_; c += 128) sx[c] = X[(size_t)row * D_ + c];
  __syncthreads();
  if (tid < 3 * V_) {
    int head = tid / V_, vi = tid % V_;
    const float* w  = (head == 0) ? hw : (head == 1) ? vw : ow;
    const float* bb = (head == 0) ? hb : (head == 1) ? vb : ob;
    float acc = bb[vi];
    for (int c = 0; c < D_; c++) acc += sx[c] * w[vi * D_ + c];
    out[(size_t)head * (B_ * T_ * V_) + (size_t)row * V_ + vi] = acc;
  }
}

// ---------------------------------------------------------------------------
extern "C" void kernel_launch(void* const* d_in, const int* in_sizes, int n_in,
                              void* d_out, int out_size, void* d_ws, size_t ws_size,
                              hipStream_t stream)
{
  (void)in_sizes; (void)n_in; (void)out_size; (void)ws_size;
  const float* tgt      = (const float*)d_in[0];
  const float* memory   = (const float*)d_in[1];
  const float* emb_w    = (const float*)d_in[2];
  const float* emb_b    = (const float*)d_in[3];
  const float* sa_qkv_w = (const float*)d_in[4];
  const float* sa_qkv_b = (const float*)d_in[5];
  const float* sa_out_w = (const float*)d_in[6];
  const float* sa_out_b = (const float*)d_in[7];
  const float* ca_qkv_w = (const float*)d_in[8];
  const float* ca_qkv_b = (const float*)d_in[9];
  const float* ca_out_w = (const float*)d_in[10];
  const float* ca_out_b = (const float*)d_in[11];
  const float* ffn_w1   = (const float*)d_in[12];
  const float* ffn_b1   = (const float*)d_in[13];
  const float* ffn_w2   = (const float*)d_in[14];
  const float* ffn_b2   = (const float*)d_in[15];
  const float* ln1_s    = (const float*)d_in[16];
  const float* ln1_b    = (const float*)d_in[17];
  const float* ln2_s    = (const float*)d_in[18];
  const float* ln2_b    = (const float*)d_in[19];
  const float* ln3_s    = (const float*)d_in[20];
  const float* ln3_b    = (const float*)d_in[21];
  const float* fn_s     = (const float*)d_in[22];
  const float* fn_b     = (const float*)d_in[23];
  const float* hit_w    = (const float*)d_in[24];
  const float* hit_b    = (const float*)d_in[25];
  const float* vel_w    = (const float*)d_in[26];
  const float* vel_b    = (const float*)d_in[27];
  const float* off_w    = (const float*)d_in[28];
  const float* off_b    = (const float*)d_in[29];

  const size_t NBT = (size_t)B_ * T_ * D_;   // 4.19M elems
  float* xf   = (float*)d_ws;
  float* tmpf = xf + NBT;
  unsigned short* xb   = (unsigned short*)(tmpf + NBT);
  unsigned short* qb   = xb + NBT;
  unsigned short* kbuf = qb + NBT;
  unsigned short* vbuf = kbuf + NBT;
  unsigned short* ab   = vbuf + NBT;                    // attention out (bf16)
  unsigned short* hbb  = ab + NBT;                      // FFN hidden (bf16)
  unsigned short* memb = hbb + (size_t)B_ * T_ * FF_;   // memory (bf16)
  unsigned short* wz   = memb + (size_t)B_ * S_ * D_;   // bf16 weight arena
  unsigned short* w_saqkv = wz;
  unsigned short* w_saout = w_saqkv + (size_t)L_ * 3 * D_ * D_;
  unsigned short* w_caqkv = w_saout + (size_t)L_ * D_ * D_;
  unsigned short* w_caout = w_caqkv + (size_t)L_ * 3 * D_ * D_;
  unsigned short* w_f1    = w_caout + (size_t)L_ * D_ * D_;
  unsigned short* w_f2    = w_f1    + (size_t)L_ * FF_ * D_;

  const int MBT = B_ * T_;   // 8192
  const int MBS = B_ * S_;   // 4096

  // ---- one-time bf16 conversion of weights + memory ----
  auto cvt = [&](const float* in, unsigned short* out, int n) {
    k_cvt<<<dim3((n + 255) / 256), dim3(256), 0, stream>>>(in, out, n);
  };
  cvt(sa_qkv_w, w_saqkv, L_ * 3 * D_ * D_);
  cvt(sa_out_w, w_saout, L_ * D_ * D_);
  cvt(ca_qkv_w, w_caqkv, L_ * 3 * D_ * D_);
  cvt(ca_out_w, w_caout, L_ * D_ * D_);
  cvt(ffn_w1,   w_f1,    L_ * FF_ * D_);
  cvt(ffn_w2,   w_f2,    L_ * D_ * FF_);
  cvt(memory,   memb,    B_ * S_ * D_);

  k_embed<<<dim3((B_ * T_ * D_ + 255) / 256), dim3(256), 0, stream>>>(tgt, emb_w, emb_b, xf, xb);

  for (int l = 0; l < L_; l++) {
    const unsigned short* wq = w_saqkv + (size_t)l * 3 * D_ * D_;
    const float* bq = sa_qkv_b + (size_t)l * 3 * D_;
    // ---- self attention ----
    k_gemm_bias<<<dim3(D_/64, MBT/128), dim3(256), 0, stream>>>(xb, wq,                    bq,        nullptr, qb,   MBT, D_, D_, 0, 1);
    k_gemm_bias<<<dim3(D_/64, MBT/128), dim3(256), 0, stream>>>(xb, wq + (size_t)D_*D_,    bq + D_,   nullptr, kbuf, MBT, D_, D_, 0, 1);
    k_gemm_bias<<<dim3(D_/64, MBT/128), dim3(256), 0, stream>>>(xb, wq + (size_t)2*D_*D_,  bq + 2*D_, nullptr, vbuf, MBT, D_, D_, 0, 1);
    k_flash_attn<<<dim3(B_*NH_, T_/32), dim3(32), 0, stream>>>(qb, kbuf, vbuf, ab, T_, T_, 0);
    k_gemm_bias<<<dim3(D_/64, MBT/128), dim3(256), 0, stream>>>(ab, w_saout + (size_t)l*D_*D_, sa_out_b + (size_t)l*D_, tmpf, nullptr, MBT, D_, D_, 0, 0);
    k_add_ln<<<dim3(MBT), dim3(256), 0, stream>>>(xf, tmpf, ln1_s + (size_t)l*D_, ln1_b + (size_t)l*D_, xf, xb, 1);
    // ---- cross attention ----
    const unsigned short* cwq = w_caqkv + (size_t)l * 3 * D_ * D_;
    const float* cbq = ca_qkv_b + (size_t)l * 3 * D_;
    k_gemm_bias<<<dim3(D_/64, MBT/128), dim3(256), 0, stream>>>(xb,   cwq,                   cbq,        nullptr, qb,   MBT, D_, D_, 0, 1);
    k_gemm_bias<<<dim3(D_/64, MBS/128), dim3(256), 0, stream>>>(memb, cwq + (size_t)D_*D_,   cbq + D_,   nullptr, kbuf, MBS, D_, D_, 0, 1);
    k_gemm_bias<<<dim3(D_/64, MBS/128), dim3(256), 0, stream>>>(memb, cwq + (size_t)2*D_*D_, cbq + 2*D_, nullptr, vbuf, MBS, D_, D_, 0, 1);
    k_flash_attn<<<dim3(B_*NH_, T_/32), dim3(32), 0, stream>>>(qb, kbuf, vbuf, ab, T_, S_, 1);
    k_gemm_bias<<<dim3(D_/64, MBT/128), dim3(256), 0, stream>>>(ab, w_caout + (size_t)l*D_*D_, ca_out_b + (size_t)l*D_, tmpf, nullptr, MBT, D_, D_, 0, 0);
    k_add_ln<<<dim3(MBT), dim3(256), 0, stream>>>(xf, tmpf, ln2_s + (size_t)l*D_, ln2_b + (size_t)l*D_, xf, xb, 1);
    // ---- FFN ----
    k_gemm_bias<<<dim3(FF_/64, MBT/128), dim3(256), 0, stream>>>(xb,  w_f1 + (size_t)l*FF_*D_, ffn_b1 + (size_t)l*FF_, nullptr, hbb, MBT, FF_, D_, 1, 1);
    k_gemm_bias<<<dim3(D_/64,  MBT/128), dim3(256), 0, stream>>>(hbb, w_f2 + (size_t)l*D_*FF_, ffn_b2 + (size_t)l*D_,  tmpf, nullptr, MBT, D_, FF_, 0, 0);
    k_add_ln<<<dim3(MBT), dim3(256), 0, stream>>>(xf, tmpf, ln3_s + (size_t)l*D_, ln3_b + (size_t)l*D_, xf, xb, 1);
  }
  k_add_ln<<<dim3(MBT), dim3(256), 0, stream>>>(xf, xf, fn_s, fn_b, xf, xb, 0);
  k_heads<<<dim3(MBT), dim3(128), 0, stream>>>(xf, hit_w, hit_b, vel_w, vel_b, off_w, off_b, (float*)d_out);
}
